// LeWinTransformerBlock_28930899706518
// MI455X (gfx1250) — matrix operand order repaired and checked by reading, not compiled, not run on hardware
//
#include <hip/hip_runtime.h>
#include <hip/hip_bf16.h>

typedef __attribute__((ext_vector_type(16))) _Float16 v16h;
typedef __attribute__((ext_vector_type(8)))  _Float16 v8h;
typedef __attribute__((ext_vector_type(8)))  float    v8f;
typedef __attribute__((ext_vector_type(4)))  int      v4i;

#define WINSZ 8
#define SHIFTSZ 4
#define NHEADS 8
#define DIMC 256
#define HIDC 1024
#define LTOT 16384

// ---- CDNA5 async global->LDS path (guarded; fallback = sync copy) ----------
#if defined(__has_builtin)
#if __has_builtin(__builtin_amdgcn_global_load_async_to_lds_b128)
#define HAVE_ASYNC_LDS 1
#endif
#endif
#ifndef HAVE_ASYNC_LDS
#define HAVE_ASYNC_LDS 0
#endif

#if HAVE_ASYNC_LDS
typedef v4i __attribute__((address_space(1))) gv4i;
typedef v4i __attribute__((address_space(3))) lv4i;
#if __has_builtin(__builtin_amdgcn_s_wait_asynccnt)
#define ASYNC_WAIT0() __builtin_amdgcn_s_wait_asynccnt(0)
#else
#define ASYNC_WAIT0() asm volatile("s_wait_asynccnt 0" ::: "memory")
#endif
#else
#define ASYNC_WAIT0()
#endif

__device__ inline void async_ld16(_Float16* lds, const _Float16* g) {
#if HAVE_ASYNC_LDS
  __builtin_amdgcn_global_load_async_to_lds_b128((gv4i*)g, (lv4i*)lds, 0, 0);
#else
  *(v8h*)lds = *(const v8h*)g;
#endif
}

__device__ inline v8f zero8() {
  v8f z;
#pragma unroll
  for (int i = 0; i < 8; i++) z[i] = 0.f;
  return z;
}

__device__ inline v16h ld_frag(const _Float16* p0, const _Float16* p1) {
  v8h lo = *(const v8h*)p0;
  v8h hi = *(const v8h*)p1;
  v16h r;
#pragma unroll
  for (int i = 0; i < 8; i++) { r[i] = lo[i]; r[i + 8] = hi[i]; }
  return r;
}

__device__ inline v8f wmma_f16(v16h a, v16h b, v8f c) {
  return __builtin_amdgcn_wmma_f32_16x16x32_f16(false, a, false, b, (short)0, c,
                                                false, false);
}

__device__ inline float gelu_f(float x) {
  return 0.5f * x * (1.0f + erff(x * 0.70710678118654752f));
}

__device__ inline int sregion(int c) { return c < 120 ? 0 : (c < 124 ? 1 : 2); }

// --------------------------------------------- weight convert + transpose
// d[n*K + k] = (f16) s[k*N + n]   (B stored N-major so GEMM tiles are
// contiguous 16B chunks -> async-loadable straight into LDS [n][k] layout)
__global__ __launch_bounds__(256) void cvt_tr_kernel(const float* __restrict__ s,
                                                     _Float16* __restrict__ d,
                                                     int K, int N) {
  int idx = blockIdx.x * 256 + threadIdx.x;
  if (idx < K * N) {
    int n = idx / K, k = idx - n * K;
    d[idx] = (_Float16)s[(size_t)k * N + n];
  }
}

// -------------------------------------------------- fused LayerNorm (+window)
__global__ __launch_bounds__(256) void ln_kernel(const float* __restrict__ x,
                                                 const float* __restrict__ g,
                                                 const float* __restrict__ b,
                                                 _Float16* __restrict__ out,
                                                 int windowed) {
  __shared__ float rs[256];
  __shared__ float rq[256];
  const int r = blockIdx.x, tid = threadIdx.x;
  size_t src;
  if (windowed) {
    const int win = r >> 6, n = r & 63;
    const int b_ = win >> 8, wi = win & 255;
    const int hh = (((wi >> 4) * 8 + (n >> 3)) + SHIFTSZ) & 127;
    const int ww = (((wi & 15) * 8 + (n & 7)) + SHIFTSZ) & 127;
    src = ((size_t)b_ * LTOT + hh * 128 + ww) * DIMC + tid;
  } else {
    src = (size_t)r * DIMC + tid;
  }
  const float v = x[src];
  rs[tid] = v;
  rq[tid] = v * v;
  __syncthreads();
  for (int s = 128; s > 0; s >>= 1) {
    if (tid < s) { rs[tid] += rs[tid + s]; rq[tid] += rq[tid + s]; }
    __syncthreads();
  }
  const float mean = rs[0] * (1.0f / 256.0f);
  const float var = rq[0] * (1.0f / 256.0f) - mean * mean;
  out[(size_t)r * DIMC + tid] =
      (_Float16)((v - mean) * rsqrtf(var + 1e-5f) * g[tid] + b[tid]);
}

// ---------------------------------------------------------- tiled WMMA GEMM
// C[M,N] = A[M,K] * Bt[N,K]^T (+epilogue).  BM=128 BN=128 BK=32, 8 waves,
// each wave a 32x64 tile (2x4 wmma / k-step), double-buffered async staging.
// MODE 0: qkv  1: proj(+win-reverse+resid)  2: lin1+gelu  3: lin2+resid
template <int MODE>
__global__ __launch_bounds__(256) void gemm_wmma_kernel(
    const _Float16* __restrict__ A, const _Float16* __restrict__ Bt,
    const float* __restrict__ bias, const float* __restrict__ resid,
    float* __restrict__ outF, _Float16* __restrict__ outH, int M, int N,
    int K) {
  __shared__ _Float16 As[2][128 * 40];  // [row][k] stride 40 halves
  __shared__ _Float16 Bs[2][128 * 40];  // [n][k]   stride 40 halves

  const int tid = threadIdx.x;
  const int wv = tid >> 5;
  const int lane = tid & 31;
  const int lh = lane >> 4;
  const int lr = lane & 15;
  const int wm = wv & 3;   // rows wm*32
  const int wn = wv >> 2;  // cols wn*64
  const int rb = blockIdx.x * 128;
  const int nb = blockIdx.y * 128;

  v8f acc[2][4];
#pragma unroll
  for (int i = 0; i < 2; i++)
#pragma unroll
    for (int j = 0; j < 4; j++) acc[i][j] = zero8();

  const int ra0 = tid >> 2;       // 0..63
  const int seg = tid & 3;        // 8-half segment
  const int so = seg * 8;

  auto stage = [&](int buf, int k0) {
    async_ld16(&As[buf][ra0 * 40 + so],
               A + (size_t)(rb + ra0) * K + k0 + so);
    async_ld16(&As[buf][(ra0 + 64) * 40 + so],
               A + (size_t)(rb + ra0 + 64) * K + k0 + so);
    async_ld16(&Bs[buf][ra0 * 40 + so],
               Bt + (size_t)(nb + ra0) * K + k0 + so);
    async_ld16(&Bs[buf][(ra0 + 64) * 40 + so],
               Bt + (size_t)(nb + ra0 + 64) * K + k0 + so);
  };

  stage(0, 0);
  ASYNC_WAIT0();
  __syncthreads();

  const int T = K >> 5;
  for (int t = 0; t < T; t++) {
    const int cur = t & 1;
    if (t + 1 < T) stage(cur ^ 1, (t + 1) << 5);

    v16h af[2], bf[4];
#pragma unroll
    for (int i = 0; i < 2; i++) {
      const _Float16* p = &As[cur][(wm * 32 + i * 16 + lr) * 40 + lh * 8];
      af[i] = ld_frag(p, p + 16);
    }
#pragma unroll
    for (int j = 0; j < 4; j++) {
      const _Float16* p = &Bs[cur][(wn * 64 + j * 16 + lr) * 40 + lh * 8];
      bf[j] = ld_frag(p, p + 16);
    }
#pragma unroll
    for (int i = 0; i < 2; i++)
#pragma unroll
      for (int j = 0; j < 4; j++) acc[i][j] = wmma_f16(af[i], bf[j], acc[i][j]);

    ASYNC_WAIT0();
    __syncthreads();
  }

#pragma unroll
  for (int i = 0; i < 2; i++) {
#pragma unroll
    for (int j = 0; j < 4; j++) {
      const int gc = nb + wn * 64 + j * 16 + lr;
#pragma unroll
      for (int r = 0; r < 8; r++) {
        const int gr = rb + wm * 32 + i * 16 + lh * 8 + r;
        float v = acc[i][j][r] + bias[gc];
        if (MODE == 0) {
          if (gc < 256) v *= 0.17677669529663687f;  // dh^-0.5
          outH[(size_t)gr * N + gc] = (_Float16)v;
        } else if (MODE == 1) {
          const int win = gr >> 6, n = gr & 63;
          const int b_ = win >> 8, wi = win & 255;
          const int hh = (((wi >> 4) * 8 + (n >> 3)) + SHIFTSZ) & 127;
          const int ww = (((wi & 15) * 8 + (n & 7)) + SHIFTSZ) & 127;
          const size_t pix = (size_t)b_ * LTOT + hh * 128 + ww;
          v += resid[pix * DIMC + gc];
          outF[pix * DIMC + gc] = v;
        } else if (MODE == 2) {
          outH[(size_t)gr * N + gc] = (_Float16)gelu_f(v);
        } else {
          v += resid[(size_t)gr * DIMC + gc];
          outF[(size_t)gr * DIMC + gc] = v;
        }
      }
    }
  }
}

// ------------------------------------------------------------- attention
__global__ __launch_bounds__(128) void attn_kernel(
    const _Float16* __restrict__ qkv, const float* __restrict__ rel_table,
    _Float16* __restrict__ outH) {
  __shared__ _Float16 qs[64 * 40];   // q[m][d]  (A layout)
  __shared__ _Float16 ks[64 * 40];   // k[m][d]  == Bs[n=m][k=d]
  __shared__ _Float16 vs[32 * 72];   // v^T[d][m] == Bs[n=d][k=m]
  __shared__ _Float16 ps[64 * 72];   // softmax(P)[n][m] (A layout, K=64)
  __shared__ float Ss[64 * 68];      // fp32 scores

  const int bid = blockIdx.x;
  const int win = bid >> 3, head = bid & 7;
  const int tid = threadIdx.x;
  const int wv = tid >> 5, lane = tid & 31;
  const int lh = lane >> 4, lr = lane & 15;
  const int wi = win & 255;
  const int nWy = wi >> 4, nWx = wi & 15;

  {  // cooperative q/k/v load (q,k async; v transposed via DS scatter)
    const int n = tid >> 1;
    const int d0 = (tid & 1) * 16;
    const _Float16* base =
        qkv + (size_t)(win * 64 + n) * 768 + head * 32 + d0;
    async_ld16(&qs[n * 40 + d0], base + 0);
    async_ld16(&qs[n * 40 + d0 + 8], base + 8);
    async_ld16(&ks[n * 40 + d0], base + 256);
    async_ld16(&ks[n * 40 + d0 + 8], base + 264);
    v8h v0 = *(const v8h*)(base + 512);
    v8h v1 = *(const v8h*)(base + 520);
#pragma unroll
    for (int j = 0; j < 8; j++) {
      vs[(d0 + j) * 72 + n] = v0[j];
      vs[(d0 + 8 + j) * 72 + n] = v1[j];
    }
  }
  ASYNC_WAIT0();
  __syncthreads();

  // S = q * k^T  (M=64,N=64,K=32); each wave a 16-row strip
  v16h afrag;
  {
    const _Float16* p = &qs[(wv * 16 + lr) * 40 + lh * 8];
    afrag = ld_frag(p, p + 16);
  }
  v8f sacc[4];
#pragma unroll
  for (int nt = 0; nt < 4; nt++) {
    sacc[nt] = zero8();
    const _Float16* p = &ks[(nt * 16 + lr) * 40 + lh * 8];
    v16h bfrag = ld_frag(p, p + 16);
    sacc[nt] = wmma_f16(afrag, bfrag, sacc[nt]);
  }

  // relative-position bias + shift mask -> fp32 scores
#pragma unroll
  for (int nt = 0; nt < 4; nt++) {
#pragma unroll
    for (int r = 0; r < 8; r++) {
      const int row = wv * 16 + lh * 8 + r;
      const int col = nt * 16 + lr;
      const int ny = row >> 3, nx = row & 7, my = col >> 3, mx = col & 7;
      const int idx = (ny - my + 7) * 15 + (nx - mx + 7);
      float b = rel_table[idx * 8 + head];
      const int gq = sregion(nWy * 8 + ny) * 3 + sregion(nWx * 8 + nx);
      const int gk = sregion(nWy * 8 + my) * 3 + sregion(nWx * 8 + mx);
      if (gq != gk) b -= 100.0f;
      Ss[row * 68 + col] = sacc[nt][r] + b;
    }
  }
  __syncthreads();

  // row softmax (fp32), write P as f16 (A layout, K=64)
  if (tid < 64) {
    float mx = -1e30f;
    for (int c = 0; c < 64; c++) mx = fmaxf(mx, Ss[tid * 68 + c]);
    float s = 0.f;
    for (int c = 0; c < 64; c++) {
      float e = __expf(Ss[tid * 68 + c] - mx);
      s += e;
      Ss[tid * 68 + c] = e;
    }
    const float inv = 1.0f / s;
    for (int c = 0; c < 64; c++)
      ps[tid * 72 + c] = (_Float16)(Ss[tid * 68 + c] * inv);
  }
  __syncthreads();

  // O = P * V  (M=64,N=32,K=64)
  v8f oacc[2];
  oacc[0] = zero8();
  oacc[1] = zero8();
#pragma unroll
  for (int kst = 0; kst < 2; kst++) {
    const _Float16* pa = &ps[(wv * 16 + lr) * 72 + kst * 32 + lh * 8];
    v16h af = ld_frag(pa, pa + 16);
#pragma unroll
    for (int nt = 0; nt < 2; nt++) {
      const _Float16* pb = &vs[(nt * 16 + lr) * 72 + kst * 32 + lh * 8];
      v16h bf = ld_frag(pb, pb + 16);
      oacc[nt] = wmma_f16(af, bf, oacc[nt]);
    }
  }
#pragma unroll
  for (int nt = 0; nt < 2; nt++) {
#pragma unroll
    for (int r = 0; r < 8; r++) {
      const int row = wv * 16 + lh * 8 + r;
      const int d = nt * 16 + lr;
      outH[(size_t)(win * 64 + row) * 256 + head * 32 + d] =
          (_Float16)oacc[nt][r];
    }
  }
}

// -------------------------------------------------- depthwise 3x3 + gelu
__global__ __launch_bounds__(256) void dwconv_kernel(
    const _Float16* __restrict__ h1, const float* __restrict__ w,
    const float* __restrict__ bias, _Float16* __restrict__ h2) {
  const int p = blockIdx.x;
  const int b_ = p >> 14, hw = p & 16383;
  const int hh = hw >> 7, ww = hw & 127;
#pragma unroll
  for (int cc = 0; cc < 4; cc++) {
    const int c = threadIdx.x + cc * 256;
    float acc = 0.f;
#pragma unroll
    for (int ky = -1; ky <= 1; ky++) {
      const int hy = hh + ky;
      if ((unsigned)hy >= 128u) continue;
#pragma unroll
      for (int kx = -1; kx <= 1; kx++) {
        const int wx = ww + kx;
        if ((unsigned)wx >= 128u) continue;
        acc += (float)h1[((size_t)b_ * LTOT + hy * 128 + wx) * HIDC + c] *
               w[c * 9 + (ky + 1) * 3 + (kx + 1)];
      }
    }
    h2[(size_t)p * HIDC + c] = (_Float16)gelu_f(acc + bias[c]);
  }
}

// ---------------------------------------------------------------- launcher
extern "C" void kernel_launch(void* const* d_in, const int* in_sizes, int n_in,
                              void* d_out, int out_size, void* d_ws,
                              size_t ws_size, hipStream_t stream) {
  (void)in_sizes; (void)n_in; (void)out_size; (void)ws_size;
  const float* x = (const float*)d_in[0];
  const float* n1g = (const float*)d_in[1];
  const float* n1b = (const float*)d_in[2];
  const float* qkv_w = (const float*)d_in[3];
  const float* qkv_b = (const float*)d_in[4];
  const float* rel_tab = (const float*)d_in[5];
  const float* proj_w = (const float*)d_in[6];
  const float* proj_b = (const float*)d_in[7];
  const float* n2g = (const float*)d_in[8];
  const float* n2b = (const float*)d_in[9];
  const float* lin1_w = (const float*)d_in[10];
  const float* lin1_b = (const float*)d_in[11];
  const float* dw_w = (const float*)d_in[12];
  const float* dw_b = (const float*)d_in[13];
  const float* lin2_w = (const float*)d_in[14];
  const float* lin2_b = (const float*)d_in[15];

  size_t off = 0;
  auto take = [&](size_t bytes) -> char* {
    char* p = (char*)d_ws + off;
    off += (bytes + 255) & ~(size_t)255;
    return p;
  };
  const size_t M = 32768;
  _Float16* xw = (_Float16*)take(M * 256 * 2);
  _Float16* qkvh = (_Float16*)take(M * 768 * 2);
  _Float16* ao = (_Float16*)take(M * 256 * 2);
  float* x1 = (float*)take(M * 256 * 4);
  _Float16* z = (_Float16*)take(M * 256 * 2);
  _Float16* h1 = (_Float16*)take(M * 1024 * 2);
  _Float16* h2 = (_Float16*)take(M * 1024 * 2);
  _Float16* qkvW = (_Float16*)take(256 * 768 * 2);    // [768][256]
  _Float16* projW = (_Float16*)take(256 * 256 * 2);   // [256][256]
  _Float16* lin1W = (_Float16*)take(256 * 1024 * 2);  // [1024][256]
  _Float16* lin2W = (_Float16*)take(1024 * 256 * 2);  // [256][1024]

  cvt_tr_kernel<<<(256 * 768 + 255) / 256, 256, 0, stream>>>(qkv_w, qkvW, 256,
                                                             768);
  cvt_tr_kernel<<<(256 * 256 + 255) / 256, 256, 0, stream>>>(proj_w, projW,
                                                             256, 256);
  cvt_tr_kernel<<<(256 * 1024 + 255) / 256, 256, 0, stream>>>(lin1_w, lin1W,
                                                              256, 1024);
  cvt_tr_kernel<<<(1024 * 256 + 255) / 256, 256, 0, stream>>>(lin2_w, lin2W,
                                                              1024, 256);

  // LN1 + roll + window partition
  ln_kernel<<<32768, 256, 0, stream>>>(x, n1g, n1b, xw, 1);
  // QKV GEMM (f16 out, q pre-scaled)
  gemm_wmma_kernel<0><<<dim3(256, 6), 256, 0, stream>>>(
      xw, qkvW, qkv_b, nullptr, nullptr, qkvh, 32768, 768, 256);
  // windowed MHA
  attn_kernel<<<4096, 128, 0, stream>>>(qkvh, rel_tab, ao);
  // proj GEMM + window-reverse + roll-back + residual -> x1 (fp32)
  gemm_wmma_kernel<1><<<dim3(256, 2), 256, 0, stream>>>(
      ao, projW, proj_b, x, x1, nullptr, 32768, 256, 256);
  // LN2
  ln_kernel<<<32768, 256, 0, stream>>>(x1, n2g, n2b, z, 0);
  // lin1 + gelu
  gemm_wmma_kernel<2><<<dim3(256, 8), 256, 0, stream>>>(
      z, lin1W, lin1_b, nullptr, nullptr, h1, 32768, 1024, 256);
  // depthwise conv + gelu
  dwconv_kernel<<<32768, 256, 0, stream>>>(h1, dw_w, dw_b, h2);
  // lin2 + residual -> output (fp32)
  gemm_wmma_kernel<3><<<dim3(256, 2), 256, 0, stream>>>(
      h2, lin2W, lin2_b, x1, (float*)d_out, nullptr, 32768, 256, 1024);
}
